// BertCounterFactTransformer_59648505806993
// MI455X (gfx1250) — matrix operand order repaired and checked by reading, not compiled, not run on hardware
//
#include <hip/hip_runtime.h>
#include <hip/hip_bf16.h>
#include <math.h>
#include <stdint.h>

#define B_ 32
#define L_ 512
#define D_ 1024
#define NEGV (-9.0e15f)
#define SCALE 0.03125f /* 1/sqrt(1024) */

typedef __bf16 bf16_t;
typedef __attribute__((ext_vector_type(16))) __bf16 v16bf;
typedef __attribute__((ext_vector_type(8)))  float  v8f;
typedef __attribute__((ext_vector_type(4)))  unsigned int u32x4;
typedef __attribute__((ext_vector_type(8)))  int i32x8;
typedef __attribute__((ext_vector_type(4)))  int i32x4;

#if __has_builtin(__builtin_amdgcn_tensor_load_to_lds) && __has_builtin(__builtin_amdgcn_s_wait_tensorcnt)
#define HAVE_TDM 1
#else
#define HAVE_TDM 0
#endif

static __device__ inline v8f zero8() {
  v8f z = {0.f,0.f,0.f,0.f,0.f,0.f,0.f,0.f};
  return z;
}

static __device__ inline v8f wmma_bf16(v16bf a, v16bf b, v8f c) {
  // D = A(16x32 bf16) x B(32x16 bf16) + C(16x16 f32)
  return __builtin_amdgcn_wmma_f32_16x16x32_bf16(false, a, false, b, (short)0, c, false, false);
}

// A-fragment (16x32 bf16), ISA layout: lane m=lane&15, K chunks at half*8 and 16+half*8.
static __device__ inline v16bf frag_a(const bf16_t* base, int ld) {
  int lane = threadIdx.x & 31;
  const bf16_t* p = base + (lane & 15) * ld + ((lane >> 4) << 3);
  union { u32x4 u[2]; v16bf v; } cv;
  cv.u[0] = *(const u32x4*)p;
  cv.u[1] = *(const u32x4*)(p + 16);
  return cv.v;
}

// B-fragment (32x16 bf16) read from a K-major [N][K] tile: lane n=lane&15, K chunks at half*8 and 16+half*8.
static __device__ inline v16bf frag_b(const bf16_t* base, int ld) {
  int lane = threadIdx.x & 31;
  const bf16_t* p = base + (lane & 15) * ld + ((lane >> 4) << 3);
  union { u32x4 u[2]; v16bf v; } cv;
  cv.u[0] = *(const u32x4*)p;
  cv.u[1] = *(const u32x4*)(p + 16);
  return cv.v;
}

// ---------------------------------------------------------------------------
// Tensor Data Mover: 2D tile load Global -> LDS (ISA ch.7/8: TENSOR_LOAD_TO_LDS)
// tile = tk elements (bf16, contiguous) x trows rows, row stride rstride elems.
// LDS padding: 4 DWORDs inserted every 16 DWORDs -> 80B LDS row pitch (LDT=40).
// ---------------------------------------------------------------------------
static __device__ inline void tdm_load_tile(unsigned lds_addr, const void* gptr,
                                            unsigned tk, unsigned trows, unsigned rstride) {
#if HAVE_TDM
  unsigned long long ga = (unsigned long long)(uintptr_t)gptr;
  // D# group0: count=1 (valid user desc), lds_addr, 57-bit global addr, type=2
  u32x4 g0 = { 1u,
               lds_addr,
               (unsigned)(ga & 0xffffffffu),
               (unsigned)((ga >> 32) & 0x01ffffffu) | (2u << 30) };
  // D# group1: wg_mask=0, data_size=1 (2B), pad_enable, pad_interval=3 (16 DW),
  //            pad_amount=3 (4 DW); tensor_dim0=tk, tensor_dim1=trows,
  //            tile_dim0=tk, tile_dim1=trows, tile_dim2=0, dim0_stride=rstride.
  int w0 = (int)((1u << 16) | (1u << 20) | (3u << 22) | (3u << 25));
  i32x8 g1 = { w0,
               (int)((tk    & 0xffffu) << 16),   // [63:48]  tensor_dim0 lo16
               (int)((trows & 0xffffu) << 16),   // [95:80]  tensor_dim1 lo16
               (int)((tk    & 0xffffu) << 16),   // [127:112] tile_dim0
               (int)(trows & 0xffffu),           // [143:128] tile_dim1
               (int)rstride,                     // [191:160] tensor_dim0_stride
               0, 0 };
  i32x4 z4 = {0, 0, 0, 0};
#if __clang_major__ >= 23
  i32x8 z8 = {0, 0, 0, 0, 0, 0, 0, 0};
  __builtin_amdgcn_tensor_load_to_lds(g0, g1, z4, z4, z8, 0);
#else
  __builtin_amdgcn_tensor_load_to_lds(g0, g1, z4, z4, 0);
#endif
#else
  (void)lds_addr; (void)gptr; (void)tk; (void)trows; (void)rstride;
#endif
}

// ---------------------------------------------------------------------------
// Kernel 1: segment masks + anomaly gate (one block per batch row)
// ---------------------------------------------------------------------------
__global__ void prep_kernel(const float* __restrict__ x, const int* __restrict__ x_ids,
                            const int* __restrict__ pad_p, const int* __restrict__ sep_p,
                            const float* __restrict__ w_anom, const float* __restrict__ b_anom,
                            float* __restrict__ gateF, float* __restrict__ fmaskF,
                            float* __restrict__ omaskF) {
  __shared__ int   ired[L_];
  __shared__ float fred[L_];
  __shared__ int   s_seppos;
  int b = blockIdx.x, l = threadIdx.x;
  int id  = x_ids[b * L_ + l];
  int pad = pad_p[0], sep = sep_p[0];
  int valid = (id != pad) ? 1 : 0;

  // first SEP position
  ired[l] = (id == sep) ? l : L_;
  __syncthreads();
  for (int s = L_ / 2; s > 0; s >>= 1) { if (l < s) ired[l] = min(ired[l], ired[l + s]); __syncthreads(); }
  int firstsep = ired[0];
  __syncthreads();
  // valid length
  ired[l] = valid;
  __syncthreads();
  for (int s = L_ / 2; s > 0; s >>= 1) { if (l < s) ired[l] += ired[l + s]; __syncthreads(); }
  int vlen = ired[0];
  __syncthreads();
  if (l == 0) {
    int fb = vlen / 2;
    fb = fb < 1 ? 1 : fb;
    fb = fb > (L_ - 2) ? (L_ - 2) : fb;
    s_seppos = (firstsep < L_) ? firstsep : fb;
  }
  __syncthreads();
  int sp = s_seppos;
  float fm = (l < sp && valid) ? 1.f : 0.f;
  float om = (l > sp && valid) ? 1.f : 0.f;
  fmaskF[b * L_ + l] = fm;
  omaskF[b * L_ + l] = om;

  // anomaly logit = x[b,l,:] . w_anom + b_anom
  const float4* xr = (const float4*)(x + ((size_t)b * L_ + l) * D_);
  const float4* wr = (const float4*)w_anom;
  float acc = 0.f;
  for (int i = 0; i < D_ / 4; ++i) {
    float4 a = xr[i], w = wr[i];
    acc += a.x * w.x + a.y * w.y + a.z * w.z + a.w * w.w;
  }
  acc += b_anom[0];

  // masked softmax over false-segment rows
  fred[l] = fm > 0.f ? acc : -INFINITY;
  __syncthreads();
  for (int s = L_ / 2; s > 0; s >>= 1) { if (l < s) fred[l] = fmaxf(fred[l], fred[l + s]); __syncthreads(); }
  float m = fred[0];
  __syncthreads();
  float e = (fm > 0.f && m != -INFINITY) ? __expf(acc - m) : 0.f;
  fred[l] = e;
  __syncthreads();
  for (int s = L_ / 2; s > 0; s >>= 1) { if (l < s) fred[l] += fred[l + s]; __syncthreads(); }
  float ssum = fred[0];
  gateF[b * L_ + l] = e / fmaxf(ssum, 1e-8f);
}

// ---------------------------------------------------------------------------
// Kernel 2: cast x fp32 -> bf16
// ---------------------------------------------------------------------------
__global__ void cast_x_kernel(const float* __restrict__ x, bf16_t* __restrict__ xb, int n) {
  int i = (blockIdx.x * blockDim.x + threadIdx.x) * 4;
  if (i < n) {
    float4 v = *(const float4*)(x + i);
    xb[i + 0] = (bf16_t)v.x;
    xb[i + 1] = (bf16_t)v.y;
    xb[i + 2] = (bf16_t)v.z;
    xb[i + 3] = (bf16_t)v.w;
  }
}

// ---------------------------------------------------------------------------
// Kernel 3: cast+transpose 6 projection weights fp32[K][N] -> bf16 W^T[N][K]
// ---------------------------------------------------------------------------
struct WPtrs { const float* w[6]; };
struct BPtrs { const float* bia[6]; };

__global__ void cast_wT_kernel(WPtrs wp, bf16_t* __restrict__ wT) {
  __shared__ float tile[32][33];
  int wi = blockIdx.z;
  const float* W = wp.w[wi];
  int k0 = blockIdx.y * 32;
  int n0 = blockIdx.x * 32;
  int tx = threadIdx.x & 31, ty = threadIdx.x >> 5;   // 256 threads: ty 0..7
  for (int r = ty; r < 32; r += 8)
    tile[r][tx] = W[(size_t)(k0 + r) * D_ + n0 + tx];
  __syncthreads();
  bf16_t* out = wT + (size_t)wi * D_ * D_;
  for (int r = ty; r < 32; r += 8)
    out[(size_t)(n0 + r) * D_ + k0 + tx] = (bf16_t)tile[tx][r];
}

// ---------------------------------------------------------------------------
// Kernel 4: six projection GEMMs  Y[w] = Xb @ W[w] + b[w]  (bf16 out)
// M=16384 N=1024 K=1024; block tile 128x128, 8 waves x (32x64) wave tiles.
// A/B tiles staged via Tensor Data Mover with double-buffered issue-ahead:
// TDM for tile k+1 is in flight (TENSORcnt) while WMMAs consume tile k.
// ---------------------------------------------------------------------------
#define LDT 40
__global__ void proj_gemm_kernel(const bf16_t* __restrict__ xb, const bf16_t* __restrict__ wT,
                                 BPtrs bp, bf16_t* __restrict__ qk) {
#if HAVE_TDM
  __shared__ bf16_t As[2][128 * LDT];
  __shared__ bf16_t Bs[2][128 * LDT];
#else
  __shared__ bf16_t As[1][128 * LDT];
  __shared__ bf16_t Bs[1][128 * LDT];
#endif
  int w  = blockIdx.z;
  int bm = blockIdx.y * 128;
  int bn = blockIdx.x * 128;
  int tid  = threadIdx.x;
  int wave = tid >> 5;
  int wm = wave & 3, wn = wave >> 2;
  const bf16_t* Wt  = wT + (size_t)w * D_ * D_;
  const float* bias = bp.bia[w];

  v8f acc[2][4];
  for (int mi = 0; mi < 2; ++mi)
    for (int ni = 0; ni < 4; ++ni) acc[mi][ni] = zero8();

#if HAVE_TDM
  unsigned ldsA[2] = { (unsigned)(uintptr_t)&As[0][0], (unsigned)(uintptr_t)&As[1][0] };
  unsigned ldsB[2] = { (unsigned)(uintptr_t)&Bs[0][0], (unsigned)(uintptr_t)&Bs[1][0] };
  // prologue: fetch tile 0 into buffer 0
  if (wave == 0) {
    tdm_load_tile(ldsA[0], &xb[(size_t)bm * D_], 32, 128, D_);
    tdm_load_tile(ldsB[0], &Wt[(size_t)bn * D_], 32, 128, D_);
    __builtin_amdgcn_s_wait_tensorcnt(0);
  }
  __syncthreads();
  int cur = 0;
  for (int kk = 0; kk < D_; kk += 32) {
    // issue-ahead: DMA tile kk+32 into the other buffer while computing this one
    if (wave == 0 && kk + 32 < D_) {
      tdm_load_tile(ldsA[cur ^ 1], &xb[(size_t)bm * D_ + kk + 32], 32, 128, D_);
      tdm_load_tile(ldsB[cur ^ 1], &Wt[(size_t)bn * D_ + kk + 32], 32, 128, D_);
    }
    v16bf af[2], bfr[4];
#pragma unroll
    for (int mi = 0; mi < 2; ++mi) af[mi] = frag_a(&As[cur][(wm * 32 + mi * 16) * LDT], LDT);
#pragma unroll
    for (int ni = 0; ni < 4; ++ni) bfr[ni] = frag_b(&Bs[cur][(wn * 64 + ni * 16) * LDT], LDT);
#pragma unroll
    for (int mi = 0; mi < 2; ++mi)
#pragma unroll
      for (int ni = 0; ni < 4; ++ni)
        acc[mi][ni] = wmma_bf16(af[mi], bfr[ni], acc[mi][ni]);
    __syncthreads();                                 // all reads of buf[cur] done
    if (wave == 0 && kk + 32 < D_) __builtin_amdgcn_s_wait_tensorcnt(0);
    __syncthreads();                                 // buf[cur^1] ready for all
    cur ^= 1;
  }
#else
  int lrow = tid >> 1, lch = (tid & 1) * 16;
  for (int kk = 0; kk < D_; kk += 32) {
    *(u32x4*)&As[0][lrow * LDT + lch] = *(const u32x4*)&xb[(size_t)(bm + lrow) * D_ + kk + lch];
    *(u32x4*)&Bs[0][lrow * LDT + lch] = *(const u32x4*)&Wt[(size_t)(bn + lrow) * D_ + kk + lch];
    if (kk + 32 < D_) {
      __builtin_prefetch(&xb[(size_t)(bm + lrow) * D_ + kk + 32 + lch], 0, 1);
      __builtin_prefetch(&Wt[(size_t)(bn + lrow) * D_ + kk + 32 + lch], 0, 1);
    }
    __syncthreads();
    v16bf af[2], bfr[4];
#pragma unroll
    for (int mi = 0; mi < 2; ++mi) af[mi] = frag_a(&As[0][(wm * 32 + mi * 16) * LDT], LDT);
#pragma unroll
    for (int ni = 0; ni < 4; ++ni) bfr[ni] = frag_b(&Bs[0][(wn * 64 + ni * 16) * LDT], LDT);
#pragma unroll
    for (int mi = 0; mi < 2; ++mi)
#pragma unroll
      for (int ni = 0; ni < 4; ++ni)
        acc[mi][ni] = wmma_bf16(af[mi], bfr[ni], acc[mi][ni]);
    __syncthreads();
  }
#endif

  int lane = tid & 31, half = lane >> 4, nl = lane & 15;
  bf16_t* out = qk + (size_t)w * (size_t)(B_ * L_) * D_;
  for (int mi = 0; mi < 2; ++mi)
    for (int ni = 0; ni < 4; ++ni) {
      int gcol = bn + wn * 64 + ni * 16 + nl;
      float bv = bias[gcol];
      int grow0 = bm + wm * 32 + mi * 16 + half * 8;
#pragma unroll
      for (int rr = 0; rr < 8; ++rr)
        out[(size_t)(grow0 + rr) * D_ + gcol] = (bf16_t)(acc[mi][ni][rr] + bv);
    }
}

// ---------------------------------------------------------------------------
// Kernel 5: attention scores + masked softmax + gate contraction.
//   flavor 0: support scores; flavor 1: repair + tanh(conflict).
// Output: gpart[b][i_tile][j] = sum_{i in tile} gate[i] * P[i,j]
// ---------------------------------------------------------------------------
#define SLD 520
__global__ void attn_gate_kernel(const bf16_t* __restrict__ qk,
                                 const float* __restrict__ gateF,
                                 const float* __restrict__ fmaskF,
                                 const float* __restrict__ omaskF,
                                 float* __restrict__ gpart_sup,
                                 float* __restrict__ gpart_rep) {
  __shared__ float S[16 * SLD];
  __shared__ float red[256];
  __shared__ float gate_s[16];
  __shared__ int   active;
  int it = blockIdx.x, b = blockIdx.y, fl = blockIdx.z;
  int tid = threadIdx.x;

  if (tid < 16) gate_s[tid] = gateF[b * L_ + it * 16 + tid];
  __syncthreads();
  if (tid == 0) {
    int a = 0;
    for (int i = 0; i < 16; ++i) a |= (gate_s[i] > 0.f);
    active = a;
  }
  __syncthreads();
  float* gpart = (fl ? gpart_rep : gpart_sup) + ((size_t)b * 32 + it) * L_;
  if (!active) { gpart[tid] = 0.f; gpart[tid + 256] = 0.f; return; }

  const size_t mat = (size_t)(B_ * L_) * D_;
  const bf16_t* qp = qk + (size_t)(fl ? 4 : 0) * mat;   // q_sup / q_rep
  const bf16_t* kp = qk + (size_t)(fl ? 5 : 1) * mat;   // k_sup / k_rep
  const bf16_t* qc = qk + 2 * mat;                      // q_con
  const bf16_t* kc = qk + 3 * mat;                      // k_con
  size_t qoff = ((size_t)b * L_ + it * 16) * D_;

  int wave = tid >> 5;
  v8f accA[4], accC[4];
  for (int j = 0; j < 4; ++j) { accA[j] = zero8(); accC[j] = zero8(); }

  for (int d = 0; d < 32; ++d) {
    v16bf a1 = frag_a(qp + qoff + d * 32, D_);
    v16bf a2 = a1;
    if (fl) a2 = frag_a(qc + qoff + d * 32, D_);
#pragma unroll
    for (int j4 = 0; j4 < 4; ++j4) {
      int jt = wave + 8 * j4;
      size_t koff = ((size_t)b * L_ + jt * 16) * D_ + d * 32;
      v16bf b1 = frag_b(kp + koff, D_);
      accA[j4] = wmma_bf16(a1, b1, accA[j4]);
      if (fl) {
        v16bf b2 = frag_b(kc + koff, D_);
        accC[j4] = wmma_bf16(a2, b2, accC[j4]);
      }
    }
  }

  int lane = tid & 31, half = lane >> 4, nl = lane & 15;
  for (int j4 = 0; j4 < 4; ++j4) {
    int j0 = (wave + 8 * j4) * 16 + nl;
#pragma unroll
    for (int rr = 0; rr < 8; ++rr) {
      float s = accA[j4][rr] * SCALE;
      float v = fl ? (s + tanhf(accC[j4][rr] * SCALE)) : s;
      S[(rr + half * 8) * SLD + j0] = v;
    }
  }
  __syncthreads();

  // masked softmax per row i; accumulate gate-weighted probability mass per column j
  float om0 = omaskF[b * L_ + tid];
  float om1 = omaskF[b * L_ + 256 + tid];
  float g0 = 0.f, g1 = 0.f;
  for (int i = 0; i < 16; ++i) {
    float gi = gate_s[i];
    if (gi <= 0.f) continue;   // uniform branch (gate_s is shared)
    float fi = fmaskF[b * L_ + it * 16 + i];
    float v0 = (fi > 0.f && om0 > 0.f) ? S[i * SLD + tid]       : NEGV;
    float v1 = (fi > 0.f && om1 > 0.f) ? S[i * SLD + 256 + tid] : NEGV;
    red[tid] = fmaxf(v0, v1); __syncthreads();
    for (int s2 = 128; s2 > 0; s2 >>= 1) { if (tid < s2) red[tid] = fmaxf(red[tid], red[tid + s2]); __syncthreads(); }
    float m = red[0]; __syncthreads();
    float e0 = __expf(v0 - m), e1 = __expf(v1 - m);
    red[tid] = e0 + e1; __syncthreads();
    for (int s2 = 128; s2 > 0; s2 >>= 1) { if (tid < s2) red[tid] += red[tid + s2]; __syncthreads(); }
    float ssum = red[0]; __syncthreads();
    float wgt = gi / ssum;
    g0 += e0 * wgt; g1 += e1 * wgt;
  }
  gpart[tid] = g0;
  gpart[tid + 256] = g1;
}

// ---------------------------------------------------------------------------
// Kernel 6: reduce partials over i-tiles -> g_sup[b,j], g_rep[b,j]
// ---------------------------------------------------------------------------
__global__ void gsum_kernel(const float* __restrict__ gps, const float* __restrict__ gpr,
                            float* __restrict__ gsupF, float* __restrict__ grepF) {
  int b = blockIdx.x, j = threadIdx.x;  // 512 threads
  float s = 0.f, r = 0.f;
  for (int it = 0; it < 32; ++it) {
    s += gps[((size_t)b * 32 + it) * L_ + j];
    r += gpr[((size_t)b * 32 + it) * L_ + j];
  }
  gsupF[b * L_ + j] = s;
  grepF[b * L_ + j] = r;
}

// ---------------------------------------------------------------------------
// Kernel 7: fused = [gate.x, grep.x, gsup.x]  ([B, 3D])
// ---------------------------------------------------------------------------
__global__ void fuse_kernel(const float* __restrict__ x, const float* __restrict__ gateF,
                            const float* __restrict__ grepF, const float* __restrict__ gsupF,
                            float* __restrict__ fused) {
  __shared__ float g0[L_], g1[L_], g2[L_];
  int b = blockIdx.x, tid = threadIdx.x;
  for (int j = tid; j < L_; j += 256) {
    g0[j] = gateF[b * L_ + j];
    g1[j] = grepF[b * L_ + j];
    g2[j] = gsupF[b * L_ + j];
  }
  __syncthreads();
  for (int q = 0; q < 4; ++q) {
    int d = q * 256 + tid;
    float s0 = 0.f, s1 = 0.f, s2 = 0.f;
    for (int j = 0; j < L_; ++j) {
      float xv = x[((size_t)b * L_ + j) * D_ + d];
      s0 += g0[j] * xv; s1 += g1[j] * xv; s2 += g2[j] * xv;
    }
    fused[(size_t)b * 3 * D_ + d]           = s0;
    fused[(size_t)b * 3 * D_ + D_ + d]      = s1;
    fused[(size_t)b * 3 * D_ + 2 * D_ + d]  = s2;
  }
}

// ---------------------------------------------------------------------------
// Kernel 8/9: FFN (relu(fused@w_f1+b1)@w_f2+b2) + LayerNorm
// ---------------------------------------------------------------------------
__global__ void ffn1_kernel(const float* __restrict__ fused, const float* __restrict__ w_f1,
                            const float* __restrict__ b_f1, float* __restrict__ hidden) {
  int b = blockIdx.y;
  int n = blockIdx.x * 256 + threadIdx.x;
  float acc = b_f1[n];
  const float* fr = fused + (size_t)b * 3 * D_;
  for (int k = 0; k < 3 * D_; ++k)
    acc = fmaf(fr[k], w_f1[(size_t)k * D_ + n], acc);
  hidden[(size_t)b * D_ + n] = fmaxf(acc, 0.f);
}

__global__ void ffn2_ln_kernel(const float* __restrict__ hidden, const float* __restrict__ w_f2,
                               const float* __restrict__ b_f2, const float* __restrict__ ln_g,
                               const float* __restrict__ ln_b, float* __restrict__ out) {
  __shared__ float val[D_];
  __shared__ float red[256];
  int b = blockIdx.x, tid = threadIdx.x;
  const float* hr = hidden + (size_t)b * D_;
  for (int q = 0; q < 4; ++q) {
    int n = q * 256 + tid;
    float a = b_f2[n];
    for (int k = 0; k < D_; ++k)
      a = fmaf(hr[k], w_f2[(size_t)k * D_ + n], a);
    val[n] = a;
  }
  __syncthreads();
  float p = 0.f;
  for (int q = 0; q < 4; ++q) p += val[q * 256 + tid];
  red[tid] = p; __syncthreads();
  for (int s = 128; s > 0; s >>= 1) { if (tid < s) red[tid] += red[tid + s]; __syncthreads(); }
  float mu = red[0] / (float)D_; __syncthreads();
  p = 0.f;
  for (int q = 0; q < 4; ++q) { float dv = val[q * 256 + tid] - mu; p += dv * dv; }
  red[tid] = p; __syncthreads();
  for (int s = 128; s > 0; s >>= 1) { if (tid < s) red[tid] += red[tid + s]; __syncthreads(); }
  float var = red[0] / (float)D_;
  float rstd = rsqrtf(var + 1e-5f);
  for (int q = 0; q < 4; ++q) {
    int n = q * 256 + tid;
    out[(size_t)b * D_ + n] = (val[n] - mu) * rstd * ln_g[n] + ln_b[n];
  }
}

// ---------------------------------------------------------------------------
extern "C" void kernel_launch(void* const* d_in, const int* in_sizes, int n_in,
                              void* d_out, int out_size, void* d_ws, size_t ws_size,
                              hipStream_t stream) {
  const float* x      = (const float*)d_in[0];
  const int*   x_ids  = (const int*)d_in[1];
  const int*   pad    = (const int*)d_in[2];
  const int*   sep    = (const int*)d_in[3];
  const float* w_anom = (const float*)d_in[4];
  const float* b_anom = (const float*)d_in[5];
  WPtrs wp; BPtrs bp;
  for (int i = 0; i < 6; ++i) {
    wp.w[i]   = (const float*)d_in[6 + 2 * i];
    bp.bia[i] = (const float*)d_in[7 + 2 * i];
  }
  const float* w_f1 = (const float*)d_in[18];
  const float* b_f1 = (const float*)d_in[19];
  const float* w_f2 = (const float*)d_in[20];
  const float* b_f2 = (const float*)d_in[21];
  const float* ln_g = (const float*)d_in[22];
  const float* ln_b = (const float*)d_in[23];
  float* out = (float*)d_out;

  char* ws = (char*)d_ws;
  size_t off = 0;
  auto alloc = [&](size_t bytes) -> void* {
    void* p = ws + off;
    off = (off + bytes + 255) & ~(size_t)255;
    return p;
  };
  bf16_t* xb    = (bf16_t*)alloc((size_t)B_ * L_ * D_ * 2);
  bf16_t* wT    = (bf16_t*)alloc((size_t)6 * D_ * D_ * 2);
  bf16_t* qk    = (bf16_t*)alloc((size_t)6 * B_ * L_ * D_ * 2);
  float* gateF  = (float*)alloc((size_t)B_ * L_ * 4);
  float* fmaskF = (float*)alloc((size_t)B_ * L_ * 4);
  float* omaskF = (float*)alloc((size_t)B_ * L_ * 4);
  float* gps    = (float*)alloc((size_t)B_ * 32 * L_ * 4);
  float* gpr    = (float*)alloc((size_t)B_ * 32 * L_ * 4);
  float* gsupF  = (float*)alloc((size_t)B_ * L_ * 4);
  float* grepF  = (float*)alloc((size_t)B_ * L_ * 4);
  float* fused  = (float*)alloc((size_t)B_ * 3 * D_ * 4);
  float* hidden = (float*)alloc((size_t)B_ * D_ * 4);
  (void)in_sizes; (void)n_in; (void)out_size; (void)ws_size;

  prep_kernel<<<B_, L_, 0, stream>>>(x, x_ids, pad, sep, w_anom, b_anom, gateF, fmaskF, omaskF);
  cast_x_kernel<<<(B_ * L_ * D_ / 4 + 255) / 256, 256, 0, stream>>>(x, xb, B_ * L_ * D_);
  cast_wT_kernel<<<dim3(D_ / 32, D_ / 32, 6), 256, 0, stream>>>(wp, wT);
  proj_gemm_kernel<<<dim3(D_ / 128, (B_ * L_) / 128, 6), 256, 0, stream>>>(xb, wT, bp, qk);
  attn_gate_kernel<<<dim3(L_ / 16, B_, 2), 256, 0, stream>>>(qk, gateF, fmaskF, omaskF, gps, gpr);
  gsum_kernel<<<B_, L_, 0, stream>>>(gps, gpr, gsupF, grepF);
  fuse_kernel<<<B_, 256, 0, stream>>>(x, gateF, grepF, gsupF, fused);
  ffn1_kernel<<<dim3(D_ / 256, B_), 256, 0, stream>>>(fused, w_f1, b_f1, hidden);
  ffn2_ln_kernel<<<B_, 256, 0, stream>>>(hidden, w_f2, b_f2, ln_g, ln_b, out);
}
